// MultiHeadedAttention_22419729285826
// MI455X (gfx1250) — compile-verified, matrix operand order checked by reading, not run on hardware
//
#include <hip/hip_runtime.h>
#include <hip/hip_bf16.h>
#include <math.h>

// ---------------------------------------------------------------------------
// MultiHeadedAttention for MI455X (gfx1250): bf16 WMMA everywhere, f32 accum.
// B=2, S=2048, D=1024, H=16, DH=64.  M = B*S = 4096.
// Software-pipelined fragment loads + TDM (tensor_load_to_lds) for V tiles.
// ---------------------------------------------------------------------------

#define BB 2
#define SS 2048
#define DD 1024
#define HH 16
#define DH 64
#define MM (BB * SS) // 4096

typedef __attribute__((ext_vector_type(16))) __bf16 v16bf;
typedef __attribute__((ext_vector_type(8)))  __bf16 v8bf;
typedef __attribute__((ext_vector_type(4)))  __bf16 v4bf;
typedef __attribute__((ext_vector_type(8)))  float  v8f;
typedef __attribute__((ext_vector_type(4)))  unsigned int v4u;
typedef __attribute__((ext_vector_type(4)))  int v4i;
typedef __attribute__((ext_vector_type(8)))  int v8i;

#define WMMA_BF16(a, b, c) \
  __builtin_amdgcn_wmma_f32_16x16x32_bf16(false, (a), false, (b), (short)0, (c), false, false)

// 16-bit WMMA fragment: per-lane = two contiguous 16B chunks, 16 elements apart.
static __device__ __forceinline__ v16bf ldfrag(const __bf16* p) {
  union { v16bf v; v8bf h[2]; } f;
  f.h[0] = *(const v8bf*)(p);
  f.h[1] = *(const v8bf*)(p + 16);
  return f.v;
}

static __device__ __forceinline__ v8f vzero8() {
  v8f z = {0.f, 0.f, 0.f, 0.f, 0.f, 0.f, 0.f, 0.f};
  return z;
}

// ---------------------------------------------------------------------------
// Tensor Data Mover: async 2D tile load Global -> LDS (CDNA5 TDM, D# per ISA §8)
// ---------------------------------------------------------------------------
#if __has_builtin(__builtin_amdgcn_tensor_load_to_lds)
#define HAVE_TDM 1
static __device__ __forceinline__ void tdm_load_2d_bf16(
    unsigned lds_off, const void* gaddr,
    unsigned tensor_d0, unsigned tensor_d1,
    unsigned tile_d0, unsigned tile_d1,
    unsigned long long stride0 /* dim0 stride, data_size units */) {
  unsigned long long ga = (unsigned long long)(size_t)gaddr;
  v4u g0;
  g0[0] = 0x1u;                                            // count=1, user D#
  g0[1] = lds_off;                                         // lds_addr (bytes)
  g0[2] = (unsigned)(ga & 0xffffffffu);                    // global_addr[31:0]
  g0[3] = (unsigned)((ga >> 32) & 0x01ffffffu) | (2u << 30); // addr[56:32] | type=2
  v8i g1;
  g1[0] = (int)(1u << 16);                                 // data_size=1 (2B), mask=0
  g1[1] = (int)((tensor_d0 & 0xffffu) << 16);              // tensor_dim0[15:0]
  g1[2] = (int)((tensor_d0 >> 16) | ((tensor_d1 & 0xffffu) << 16)); // d0 hi | d1 lo
  g1[3] = (int)((tensor_d1 >> 16) | (tile_d0 << 16));      // d1 hi | tile_dim0
  g1[4] = (int)(tile_d1 & 0xffffu);                        // tile_dim1 (tile_dim2=0)
  g1[5] = (int)(unsigned)(stride0 & 0xffffffffull);        // dim0_stride[31:0]
  g1[6] = (int)(unsigned)((stride0 >> 32) & 0xffffull);    // dim0_stride[47:32]
  g1[7] = 0;
  v4i gz = {0, 0, 0, 0};
#if defined(__clang_major__) && (__clang_major__ >= 23)
  v8i gz8 = {0, 0, 0, 0, 0, 0, 0, 0};
  __builtin_amdgcn_tensor_load_to_lds(g0, g1, gz, gz, gz8, 0);
#else
  __builtin_amdgcn_tensor_load_to_lds(g0, g1, gz, gz, 0);
#endif
}
static __device__ __forceinline__ void wait_tensor0() {
#if __has_builtin(__builtin_amdgcn_s_wait_tensorcnt)
  __builtin_amdgcn_s_wait_tensorcnt(0);
#else
  asm volatile("s_wait_tensorcnt 0x0" ::: "memory");
#endif
}
#else
#define HAVE_TDM 0
#endif

// ---------------------------------------------------------------------------
// Kernel 0: f32 -> bf16 conversion (query, weights).  n % 4 == 0.
// ---------------------------------------------------------------------------
__global__ void __launch_bounds__(256) cvt_kernel(const float* __restrict__ in,
                                                  __bf16* __restrict__ out, int n) {
  int i = (blockIdx.x * 256 + threadIdx.x) * 4;
  if (i + 3 < n) {
    float4 f = *(const float4*)(in + i);
    v4bf o;
    o[0] = (__bf16)f.x; o[1] = (__bf16)f.y; o[2] = (__bf16)f.z; o[3] = (__bf16)f.w;
    *(v4bf*)(out + i) = o;
  }
}

// ---------------------------------------------------------------------------
// Kernel 1: fused projection  Y = X @ W^T + bias
// mode 0: Q -> [B,H,S,DH], scaled by 1/sqrt(DH)
// mode 1: K -> [B,H,S,DH]
// mode 2: V -> [B,H,DH,S]   (transposed for contiguous PV B-fragments)
// One wave computes a 16x64 strip; double-buffered fragment pipeline.
// Grid: 4096 waves = 256 M-tiles * 16 N-blocks.
// ---------------------------------------------------------------------------
__global__ void __launch_bounds__(256) proj_kernel(const __bf16* __restrict__ X,
                                                   const __bf16* __restrict__ W,
                                                   const float*  __restrict__ bias,
                                                   __bf16* __restrict__ out,
                                                   int mode) {
  const int wid  = (blockIdx.x * 256 + threadIdx.x) >> 5;
  const int lane = threadIdx.x & 31;
  const int mt = wid >> 4;        // 0..255
  const int nb = wid & 15;        // 0..15
  const int m0 = mt * 16;
  const int n0 = nb * 64;
  const int lm = lane & 15;
  const int hi = lane >> 4;       // lane half selects K-chunk 0 / 8
  const int kc = hi * 8;

  v8f acc[4] = {vzero8(), vzero8(), vzero8(), vzero8()};

  const __bf16* arow  = X + (size_t)(m0 + lm) * DD + kc;
  const __bf16* brow0 = W + (size_t)(n0 +  0 + lm) * DD + kc;
  const __bf16* brow1 = W + (size_t)(n0 + 16 + lm) * DD + kc;
  const __bf16* brow2 = W + (size_t)(n0 + 32 + lm) * DD + kc;
  const __bf16* brow3 = W + (size_t)(n0 + 48 + lm) * DD + kc;

  // ---- pipelined main loop: loads of step k+32 issue before WMMAs of step k
  v16bf a  = ldfrag(arow);
  v16bf b0 = ldfrag(brow0);
  v16bf b1 = ldfrag(brow1);
  v16bf b2 = ldfrag(brow2);
  v16bf b3 = ldfrag(brow3);
  for (int k0 = 32; k0 < DD; k0 += 32) {
    v16bf an  = ldfrag(arow + k0);
    v16bf bn0 = ldfrag(brow0 + k0);
    v16bf bn1 = ldfrag(brow1 + k0);
    v16bf bn2 = ldfrag(brow2 + k0);
    v16bf bn3 = ldfrag(brow3 + k0);
    acc[0] = WMMA_BF16(a, b0, acc[0]);
    acc[1] = WMMA_BF16(a, b1, acc[1]);
    acc[2] = WMMA_BF16(a, b2, acc[2]);
    acc[3] = WMMA_BF16(a, b3, acc[3]);
    a = an; b0 = bn0; b1 = bn1; b2 = bn2; b3 = bn3;
  }
  acc[0] = WMMA_BF16(a, b0, acc[0]);
  acc[1] = WMMA_BF16(a, b1, acc[1]);
  acc[2] = WMMA_BF16(a, b2, acc[2]);
  acc[3] = WMMA_BF16(a, b3, acc[3]);

#pragma unroll
  for (int t = 0; t < 4; ++t) {
    const int n  = n0 + t * 16 + lm;     // C layout: N = lane%16
    const float bv = bias[n];
    const int h  = n >> 6;
    const int dh = n & 63;
#pragma unroll
    for (int r = 0; r < 8; ++r) {
      const int m = m0 + r + hi * 8;     // lanes 16-31 hold rows r+8
      const int b = m >> 11;             // / S
      const int s = m & (SS - 1);
      float v = acc[t][r] + bv;
      if (mode == 0) {
        out[(((size_t)b * HH + h) * SS + s) * DH + dh] = (__bf16)(v * 0.125f);
      } else if (mode == 1) {
        out[(((size_t)b * HH + h) * SS + s) * DH + dh] = (__bf16)v;
      } else {
        out[(((size_t)b * HH + h) * DH + dh) * SS + s] = (__bf16)v;
      }
    }
  }
}

// ---------------------------------------------------------------------------
// Kernel 2: flash attention.  One wave per (b,h,16-query tile).
// Q:[B,H,S,DH] (pre-scaled), K:[B,H,S,DH], Vt:[B,H,DH,S], mask:[B,S,S] bool,
// ctx out: [B,S,D] bf16.
// K fragments double-buffered across key blocks (global->VGPR, pipelined).
// V tiles fetched by the Tensor Data Mover into a per-wave LDS slot (async,
// TENSORcnt), consumed after softmax; falls back to direct loads without TDM.
// Grid: 4096 waves = 32 (b,h) * 128 q-tiles.
// ---------------------------------------------------------------------------
__global__ void __launch_bounds__(256) attn_kernel(const __bf16* __restrict__ Q,
                                                   const __bf16* __restrict__ Kmat,
                                                   const __bf16* __restrict__ Vt,
                                                   const unsigned char* __restrict__ mask,
                                                   __bf16* __restrict__ ctx) {
  __shared__ __bf16 Plds[8][16 * 32];   // per-wave 16x32 probability tile (1KB/wave)
#if HAVE_TDM
  __shared__ __bf16 Vlds[8][64 * 32];   // per-wave 64(dh) x 32(key) V tile (4KB/wave)
#endif

  const int wid   = (blockIdx.x * 256 + threadIdx.x) >> 5;
  const int lane  = threadIdx.x & 31;
  const int wslot = (threadIdx.x >> 5);
  const int qt = wid & 127;            // q tile within (b,h)
  const int bh = wid >> 7;             // 0..31
  const int b  = bh >> 4;
  const int h  = bh & 15;
  const int q0 = qt * 16;
  const int lm = lane & 15;
  const int hi = lane >> 4;
  const int kc = hi * 8;

  // Q fragments for the two K-steps over DH (dh 0..31, 32..63)
  const __bf16* qbase = Q + ((size_t)bh * SS + q0 + lm) * DH + kc;
  v16bf qa0 = ldfrag(qbase);
  v16bf qa1 = ldfrag(qbase + 32);

  const __bf16* kbase = Kmat + (size_t)bh * SS * DH;
  const __bf16* vbase = Vt + (size_t)bh * DH * SS;
  const unsigned char* mbase = mask + (size_t)b * SS * SS;

#if HAVE_TDM
  const unsigned vslot_off = (unsigned)(size_t)(void*)&Vlds[wslot][0];
#endif

  v8f o[4] = {vzero8(), vzero8(), vzero8(), vzero8()};
  float mx[8], lsum[8];
#pragma unroll
  for (int r = 0; r < 8; ++r) { mx[r] = -3.0e38f; lsum[r] = 0.f; }

  // preload K fragments for key block 0
  v16bf kf0, kf1, kf2, kf3;
  {
    const __bf16* kr0 = kbase + (size_t)(lm) * DH + kc;
    const __bf16* kr1 = kbase + (size_t)(16 + lm) * DH + kc;
    kf0 = ldfrag(kr0); kf1 = ldfrag(kr0 + 32);
    kf2 = ldfrag(kr1); kf3 = ldfrag(kr1 + 32);
  }

  for (int kb = 0; kb < SS; kb += 32) {
#if HAVE_TDM
    // ---- async TDM: 2D tile (32 keys x 64 dh rows, row stride S) -> LDS slot.
    //      Issued first; consumed after QK^T + softmax + P bounce.
    tdm_load_2d_bf16(vslot_off, (const void*)(vbase + kb),
                     /*tensor_d0=*/SS, /*tensor_d1=*/DH,
                     /*tile_d0=*/32, /*tile_d1=*/DH,
                     /*stride0=*/(unsigned long long)SS);
#else
    const __bf16* vr = vbase + (size_t)lm * SS + kb + kc;
    v16bf vf0 = ldfrag(vr);
    v16bf vf1 = ldfrag(vr + (size_t)16 * SS);
    v16bf vf2 = ldfrag(vr + (size_t)32 * SS);
    v16bf vf3 = ldfrag(vr + (size_t)48 * SS);
#endif

    // ---- scores S = Q K^T for this 16x32 key block
    v8f sc0 = vzero8(), sc1 = vzero8();
    sc0 = WMMA_BF16(qa0, kf0, sc0);
    sc0 = WMMA_BF16(qa1, kf1, sc0);
    sc1 = WMMA_BF16(qa0, kf2, sc1);
    sc1 = WMMA_BF16(qa1, kf3, sc1);

    // ---- prefetch next block's K fragments (latency hidden by softmax)
    v16bf kn0 = kf0, kn1 = kf1, kn2 = kf2, kn3 = kf3;
    if (kb + 32 < SS) {
      const __bf16* kr0 = kbase + (size_t)(kb + 32 + lm) * DH + kc;
      const __bf16* kr1 = kbase + (size_t)(kb + 48 + lm) * DH + kc;
      kn0 = ldfrag(kr0); kn1 = ldfrag(kr0 + 32);
      kn2 = ldfrag(kr1); kn3 = ldfrag(kr1 + 32);
    }

    // ---- mask bytes (batched loads, then apply):  score = mask ? -1e18 : score
    unsigned char mb0[8], mb1[8];
#pragma unroll
    for (int r = 0; r < 8; ++r) {
      const unsigned char* mp = mbase + (size_t)(q0 + r + hi * 8) * SS + kb;
      mb0[r] = mp[lm];
      mb1[r] = mp[16 + lm];
    }
#pragma unroll
    for (int r = 0; r < 8; ++r) {
      if (mb0[r]) sc0[r] = -1.0e18f;
      if (mb1[r]) sc1[r] = -1.0e18f;
    }

    // ---- online softmax (rows live across 16-lane halves; reduce with xor 1..8)
#pragma unroll
    for (int r = 0; r < 8; ++r) {
      float bm = fmaxf(sc0[r], sc1[r]);
      bm = fmaxf(bm, __shfl_xor(bm, 1, 32));
      bm = fmaxf(bm, __shfl_xor(bm, 2, 32));
      bm = fmaxf(bm, __shfl_xor(bm, 4, 32));
      bm = fmaxf(bm, __shfl_xor(bm, 8, 32));
      const float mnew  = fmaxf(mx[r], bm);
      const float alpha = __expf(mx[r] - mnew);
      const float e0 = __expf(sc0[r] - mnew);
      const float e1 = __expf(sc1[r] - mnew);
      float rs = e0 + e1;
      rs += __shfl_xor(rs, 1, 32);
      rs += __shfl_xor(rs, 2, 32);
      rs += __shfl_xor(rs, 4, 32);
      rs += __shfl_xor(rs, 8, 32);
      lsum[r] = lsum[r] * alpha + rs;
      mx[r] = mnew;
      o[0][r] *= alpha; o[1][r] *= alpha; o[2][r] *= alpha; o[3][r] *= alpha;
      // P tile to LDS in [q][key] order (C layout -> A layout redistribution)
      const int row = r + hi * 8;
      Plds[wslot][row * 32 + lm]      = (__bf16)e0;
      Plds[wslot][row * 32 + 16 + lm] = (__bf16)e1;
    }

    asm volatile("s_wait_dscnt 0x0" ::: "memory");  // LDS RAW within the wave

    // ---- A fragment of P (16 q x 32 keys) read back in WMMA A layout
    const __bf16* pl = &Plds[wslot][lm * 32 + kc];
    v16bf pa = ldfrag(pl);

#if HAVE_TDM
    // ---- V fragments from the TDM-staged LDS tile
    wait_tensor0();
    v16bf vf0 = ldfrag(&Vlds[wslot][( 0 + lm) * 32 + kc]);
    v16bf vf1 = ldfrag(&Vlds[wslot][(16 + lm) * 32 + kc]);
    v16bf vf2 = ldfrag(&Vlds[wslot][(32 + lm) * 32 + kc]);
    v16bf vf3 = ldfrag(&Vlds[wslot][(48 + lm) * 32 + kc]);
#endif

    // ---- O += P @ V   (4 dh N-tiles)
    o[0] = WMMA_BF16(pa, vf0, o[0]);
    o[1] = WMMA_BF16(pa, vf1, o[1]);
    o[2] = WMMA_BF16(pa, vf2, o[2]);
    o[3] = WMMA_BF16(pa, vf3, o[3]);

    kf0 = kn0; kf1 = kn1; kf2 = kn2; kf3 = kn3;
  }

  // ---- normalize and store ctx[b, q, h*64 + dh] as bf16
  float inv[8];
#pragma unroll
  for (int r = 0; r < 8; ++r) inv[r] = (lsum[r] > 0.f) ? (1.0f / lsum[r]) : 0.f;
#pragma unroll
  for (int t = 0; t < 4; ++t) {
#pragma unroll
    for (int r = 0; r < 8; ++r) {
      const int q = q0 + r + hi * 8;
      ctx[((size_t)b * SS + q) * DD + h * DH + t * 16 + lm] = (__bf16)(o[t][r] * inv[r]);
    }
  }
}

// ---------------------------------------------------------------------------
// Kernel 3: output projection  out = ctx @ Wo^T + bo   (f32 stores to d_out)
// ---------------------------------------------------------------------------
__global__ void __launch_bounds__(256) oproj_kernel(const __bf16* __restrict__ X,
                                                    const __bf16* __restrict__ W,
                                                    const float*  __restrict__ bias,
                                                    float* __restrict__ out) {
  const int wid  = (blockIdx.x * 256 + threadIdx.x) >> 5;
  const int lane = threadIdx.x & 31;
  const int mt = wid >> 4;
  const int nb = wid & 15;
  const int m0 = mt * 16;
  const int n0 = nb * 64;
  const int lm = lane & 15;
  const int hi = lane >> 4;
  const int kc = hi * 8;

  v8f acc[4] = {vzero8(), vzero8(), vzero8(), vzero8()};

  const __bf16* arow  = X + (size_t)(m0 + lm) * DD + kc;
  const __bf16* brow0 = W + (size_t)(n0 +  0 + lm) * DD + kc;
  const __bf16* brow1 = W + (size_t)(n0 + 16 + lm) * DD + kc;
  const __bf16* brow2 = W + (size_t)(n0 + 32 + lm) * DD + kc;
  const __bf16* brow3 = W + (size_t)(n0 + 48 + lm) * DD + kc;

  v16bf a  = ldfrag(arow);
  v16bf b0 = ldfrag(brow0);
  v16bf b1 = ldfrag(brow1);
  v16bf b2 = ldfrag(brow2);
  v16bf b3 = ldfrag(brow3);
  for (int k0 = 32; k0 < DD; k0 += 32) {
    v16bf an  = ldfrag(arow + k0);
    v16bf bn0 = ldfrag(brow0 + k0);
    v16bf bn1 = ldfrag(brow1 + k0);
    v16bf bn2 = ldfrag(brow2 + k0);
    v16bf bn3 = ldfrag(brow3 + k0);
    acc[0] = WMMA_BF16(a, b0, acc[0]);
    acc[1] = WMMA_BF16(a, b1, acc[1]);
    acc[2] = WMMA_BF16(a, b2, acc[2]);
    acc[3] = WMMA_BF16(a, b3, acc[3]);
    a = an; b0 = bn0; b1 = bn1; b2 = bn2; b3 = bn3;
  }
  acc[0] = WMMA_BF16(a, b0, acc[0]);
  acc[1] = WMMA_BF16(a, b1, acc[1]);
  acc[2] = WMMA_BF16(a, b2, acc[2]);
  acc[3] = WMMA_BF16(a, b3, acc[3]);

#pragma unroll
  for (int t = 0; t < 4; ++t) {
    const int n = n0 + t * 16 + lm;
    const float bv = bias[n];
#pragma unroll
    for (int r = 0; r < 8; ++r) {
      const int m = m0 + r + hi * 8;
      out[(size_t)m * DD + n] = acc[t][r] + bv;
    }
  }
}

// ---------------------------------------------------------------------------
// launch
// ---------------------------------------------------------------------------
extern "C" void kernel_launch(void* const* d_in, const int* in_sizes, int n_in,
                              void* d_out, int out_size, void* d_ws, size_t ws_size,
                              hipStream_t stream) {
  const float* query        = (const float*)d_in[0];
  const unsigned char* mask = (const unsigned char*)d_in[1]; // jax bool = 1 byte
  const float* Wq = (const float*)d_in[2];
  const float* bq = (const float*)d_in[3];
  const float* Wk = (const float*)d_in[4];
  const float* bk = (const float*)d_in[5];
  const float* Wv = (const float*)d_in[6];
  const float* bv = (const float*)d_in[7];
  const float* Wo = (const float*)d_in[8];
  const float* bo = (const float*)d_in[9];

  char* ws = (char*)d_ws;
  // workspace layout (bf16 staging, ~48 MB total)
  __bf16* qbf = (__bf16*)(ws);                        //  8 MB  [M,D]
  __bf16* wqb = (__bf16*)(ws + (8u << 20));           //  2 MB
  __bf16* wkb = (__bf16*)(ws + (10u << 20));          //  2 MB
  __bf16* wvb = (__bf16*)(ws + (12u << 20));          //  2 MB
  __bf16* wob = (__bf16*)(ws + (14u << 20));          //  2 MB
  __bf16* Qw  = (__bf16*)(ws + (16u << 20));          //  8 MB  [B,H,S,DH]
  __bf16* Kw  = (__bf16*)(ws + (24u << 20));          //  8 MB  [B,H,S,DH]
  __bf16* Vtw = (__bf16*)(ws + (32u << 20));          //  8 MB  [B,H,DH,S]
  __bf16* Cw  = (__bf16*)(ws + (40u << 20));          //  8 MB  [B,S,D]

  const int MD = MM * DD;      // 4,194,304
  const int WW = DD * DD;      // 1,048,576

  cvt_kernel<<<MD / 4 / 256, 256, 0, stream>>>(query, qbf, MD);
  cvt_kernel<<<WW / 4 / 256, 256, 0, stream>>>(Wq, wqb, WW);
  cvt_kernel<<<WW / 4 / 256, 256, 0, stream>>>(Wk, wkb, WW);
  cvt_kernel<<<WW / 4 / 256, 256, 0, stream>>>(Wv, wvb, WW);
  cvt_kernel<<<WW / 4 / 256, 256, 0, stream>>>(Wo, wob, WW);

  // 4096 waves each: 256 M-tiles x 16 N-blocks
  proj_kernel<<<512, 256, 0, stream>>>(qbf, wqb, bq, Qw, 0);
  proj_kernel<<<512, 256, 0, stream>>>(qbf, wkb, bk, Kw, 1);
  proj_kernel<<<512, 256, 0, stream>>>(qbf, wvb, bv, Vtw, 2);

  attn_kernel<<<512, 256, 0, stream>>>(Qw, Kw, Vtw, mask, Cw);

  oproj_kernel<<<512, 256, 0, stream>>>(Cw, wob, bo, (float*)d_out);
}
